// AttentionTopK_11914239279762
// MI455X (gfx1250) — matrix operand order repaired and checked by reading, not compiled, not run on hardware
//
#include <hip/hip_runtime.h>
#include <hip/hip_bf16.h>

typedef __attribute__((ext_vector_type(16))) __bf16 v16bf;
typedef __attribute__((ext_vector_type(8)))  __bf16 v8bf;
typedef __attribute__((ext_vector_type(8)))  float  v8f;
typedef __attribute__((ext_vector_type(4)))  int    v4i;

typedef __attribute__((address_space(1))) v4i* gas_v4i;   // global int4*
typedef __attribute__((address_space(3))) v4i* las_v4i;   // LDS int4*

#define NB 128   // batches
#define NN 512   // sequence
#define ND 256   // feature dim
#define TK 8     // top-k
#define LDT 264  // padded LDS row stride (bf16 elems) for 256-wide tiles

// ---------------------------------------------------------------------------
// CDNA5 async global->LDS copy (ASYNCcnt path, no VGPR staging)
// ---------------------------------------------------------------------------
__device__ __forceinline__ void async_ld_b128(void* lds_dst, const void* gsrc) {
    __builtin_amdgcn_global_load_async_to_lds_b128(
        (gas_v4i)(unsigned long long)gsrc,
        (las_v4i)(unsigned)(unsigned long long)lds_dst,
        /*imm offset=*/0, /*cpol=*/0);
}
__device__ __forceinline__ void wait_async0() {
    __builtin_amdgcn_s_wait_asynccnt(0);
}

// Issue async copy of a 64x256 bf16 tile (row stride 256) into padded LDS.
// 2048 16B chunks / 256 threads = 8 async ops per thread.
__device__ __forceinline__ void copy_tile_async(__bf16* dst, const __bf16* src, int t) {
#pragma unroll
    for (int i = t; i < 64 * 32; i += 256) {
        int r = i >> 5, c = i & 31;
        async_ld_b128(dst + r * LDT + c * 8, src + r * ND + c * 8);
    }
}

// ---------------------------------------------------------------------------
// WMMA helpers
// ---------------------------------------------------------------------------
__device__ __forceinline__ v8f wmma_bf16(v16bf a, v16bf b, v8f c) {
    return __builtin_amdgcn_wmma_f32_16x16x32_bf16(
        /*neg_a=*/false, a, /*neg_b=*/false, b,
        /*c_mod=*/(short)0, c, /*reuse_a=*/false, /*reuse_b=*/false);
}

// A-matrix 16x32 bf16 fragment (ISA 7.12.2): lane l (m=l&15), kb=(l>=16)?8:0,
// elems[0..7]=K kb..kb+7, elems[8..15]=K kb+16..kb+23.
__device__ __forceinline__ v16bf load_a_frag(const __bf16* tile, int row0,
                                             int k0, unsigned lane) {
    int m  = lane & 15;
    int kb = (lane >> 4) * 8;
    const __bf16* p = tile + (row0 + m) * LDT + k0 + kb;
    v8bf lo = *(const v8bf*)p;
    v8bf hi = *(const v8bf*)(p + 16);
    v16bf out;
#pragma unroll
    for (int i = 0; i < 8; ++i) { out[i] = lo[i]; out[i + 8] = hi[i]; }
    return out;
}

// B-matrix 32x16 bf16 fragment: lane l holds column n=l&15, kb=(l>=16)?8:0,
// elems[0..7]=K kb..kb+7, elems[8..15]=K kb+16..kb+23.
__device__ __forceinline__ v16bf load_b_frag(const __bf16* tile, int col0,
                                             int k0, unsigned lane) {
    int n  = lane & 15;
    int kb = (lane >> 4) * 8;
    const __bf16* p = tile + (col0 + n) * LDT + k0 + kb;
    v8bf lo = *(const v8bf*)p;
    v8bf hi = *(const v8bf*)(p + 16);
    v16bf out;
#pragma unroll
    for (int i = 0; i < 8; ++i) { out[i] = lo[i]; out[i + 8] = hi[i]; }
    return out;
}

// ---------------------------------------------------------------------------
// Prep kernels
// ---------------------------------------------------------------------------
__global__ __launch_bounds__(256) void prep_xs_kernel(
    const float* __restrict__ x, const unsigned char* __restrict__ mask,
    __bf16* __restrict__ xsb) {
    size_t i = ((size_t)blockIdx.x * 256 + threadIdx.x) * 4;
    size_t row = i / ND;                       // b*NN + n
    float m = mask[row] ? 0.0625f : 0.0f;      // mask / sqrt(256)
    float4 v = *(const float4*)(x + i);
    xsb[i + 0] = (__bf16)(v.x * m);
    xsb[i + 1] = (__bf16)(v.y * m);
    xsb[i + 2] = (__bf16)(v.z * m);
    xsb[i + 3] = (__bf16)(v.w * m);
}

__global__ __launch_bounds__(256) void prep_w_kernel(
    const float* __restrict__ W, __bf16* __restrict__ Wb) {
    size_t i = ((size_t)blockIdx.x * 256 + threadIdx.x) * 4;
    float4 v = *(const float4*)(W + i);
    Wb[i + 0] = (__bf16)v.x;
    Wb[i + 1] = (__bf16)v.y;
    Wb[i + 2] = (__bf16)v.z;
    Wb[i + 3] = (__bf16)v.w;
}

__global__ void calc_denom_kernel(const unsigned char* __restrict__ mask,
                                  float* __restrict__ denom) {
    int b = threadIdx.x;
    if (b < NB) {
        int s = 0;
        for (int n = 0; n < NN; ++n) s += mask[b * NN + n] ? 1 : 0;
        int d = s - 1;
        if (d > TK) d = TK;
        denom[b] = (float)d;
    }
}

// ---------------------------------------------------------------------------
// Phase 2: sims tiles via WMMA + fused top-8 + gather/average -> attn (bf16).
// One block per (batch, 64-row slab); B-tiles double-buffered via async LDS
// loads so the next tile's copy overlaps WMMA + top-k scan.
// ---------------------------------------------------------------------------
__global__ __launch_bounds__(256) void topk_attn_kernel(
    const float* __restrict__ x, const unsigned char* __restrict__ mask,
    const __bf16* __restrict__ xsb, const float* __restrict__ denom,
    __bf16* __restrict__ attnb) {
    __shared__ __bf16 As[64 * LDT];
    __shared__ __bf16 Bs[2][64 * LDT];
    __shared__ float  Ct[64 * 65];           // stride 65: conflict-free row scan
    __shared__ int    topIdxS[64 * TK];
    __shared__ unsigned char maskS[NN];

    const int slab = blockIdx.x;             // 0..7
    const int b    = blockIdx.y;             // batch
    const int t    = threadIdx.x;
    const unsigned lane = t & 31;
    const int wave = t >> 5;
    const int row0 = slab * 64;

    // stage A slab + first B tile asynchronously; mask row into LDS
    copy_tile_async(As, xsb + (size_t)(b * NN + row0) * ND, t);
    copy_tile_async(Bs[0], xsb + (size_t)(b * NN) * ND, t);
    if (t < 32)
        *(uint4*)&maskS[t * 16] = ((const uint4*)(mask + b * NN))[t];
    wait_async0();
    __syncthreads();

    float topV[TK];
    int   topI[TK];
#pragma unroll
    for (int j = 0; j < TK; ++j) { topV[j] = -3.0e38f; topI[j] = 0; }
    const int nGlob = row0 + t;              // row owned by thread t (<64)

    const int mr  = wave >> 1;               // A-row subtile 0..3
    const int nc0 = (wave & 1) * 2;          // first of two B-col subtiles

    for (int ct = 0; ct < 8; ++ct) {
        const __bf16* Bcur = Bs[ct & 1];
        if (ct < 7)                          // prefetch next tile (overlapped)
            copy_tile_async(Bs[(ct + 1) & 1],
                            xsb + (size_t)(b * NN + (ct + 1) * 64) * ND, t);

        v8f c0 = {}; v8f c1 = {};
#pragma unroll
        for (int ks = 0; ks < 8; ++ks) {     // K = 256 in steps of 32
            v16bf a  = load_a_frag(As, mr * 16, ks * 32, lane);
            v16bf b0 = load_b_frag(Bcur, nc0 * 16, ks * 32, lane);
            v16bf b1 = load_b_frag(Bcur, (nc0 + 1) * 16, ks * 32, lane);
            c0 = wmma_bf16(a, b0, c0);
            c1 = wmma_bf16(a, b1, c1);
        }

        // spill C frags to LDS (doc layout: VGPR r -> row mr*16+r (+8 hi half))
        {
            int rB   = mr * 16 + ((lane >= 16) ? 8 : 0);
            int col0 = nc0 * 16 + (lane & 15);
#pragma unroll
            for (int r = 0; r < 8; ++r) {
                Ct[(rB + r) * 65 + col0]      = c0[r];
                Ct[(rB + r) * 65 + col0 + 16] = c1[r];
            }
        }
        __syncthreads();                     // Ct visible to scan threads

        // top-8 insertion scan: thread t<64 owns sims row t
        if (t < 64) {
#pragma unroll 1
            for (int c = 0; c < 64; ++c) {
                int colg = ct * 64 + c;
                if (colg == nGlob) continue;             // diagonal -> -inf
                float v = maskS[colg] ? Ct[t * 65 + c] : -100.0f;
                if (v > topV[TK - 1]) {
                    int j = TK - 1;
                    while (j > 0 && topV[j - 1] < v) {
                        topV[j] = topV[j - 1]; topI[j] = topI[j - 1]; --j;
                    }
                    topV[j] = v; topI[j] = colg;
                }
            }
        }
        if (ct < 7) wait_async0();           // own async writes for next buffer
        __syncthreads();                     // everyone's next buffer + scan done
    }

    if (t < 64) {
#pragma unroll
        for (int j = 0; j < TK; ++j) topIdxS[t * TK + j] = topI[j];
    }
    __syncthreads();

    // gather + average: 4 threads per row, 64 features each (x is L2-resident)
    const float invDn = 1.0f / denom[b];
    const int r    = t >> 2;
    const int dseg = (t & 3) * 64;
    const int n    = row0 + r;
    for (int dblk = 0; dblk < 4; ++dblk) {
        const int d0 = dseg + dblk * 16;
        float acc[16];
#pragma unroll
        for (int q = 0; q < 16; ++q) acc[q] = 0.0f;
#pragma unroll
        for (int j = 0; j < TK; ++j) {
            int idx = topIdxS[r * TK + j];
            float s = maskS[idx] ? 0.0625f : 0.0f;       // xs = x*mask/16
            const float4* p = (const float4*)(x + ((size_t)(b * NN + idx) * ND + d0));
#pragma unroll
            for (int q = 0; q < 4; ++q) {
                float4 v = p[q];
                acc[q * 4 + 0] += v.x * s;
                acc[q * 4 + 1] += v.y * s;
                acc[q * 4 + 2] += v.z * s;
                acc[q * 4 + 3] += v.w * s;
            }
        }
        __bf16* dst = attnb + ((size_t)(b * NN + n) * ND + d0);
#pragma unroll
        for (int q = 0; q < 16; ++q) dst[q] = (__bf16)(acc[q] * invDn);
    }
}

// ---------------------------------------------------------------------------
// Phase 3: out[i, n] = sum_k attn[i,k] * W[n,k] + bias[n]   (WMMA bf16)
// ---------------------------------------------------------------------------
__global__ __launch_bounds__(256) void final_gemm_kernel(
    const __bf16* __restrict__ attnb, const __bf16* __restrict__ Wb,
    const float* __restrict__ bias, float* __restrict__ out) {
    __shared__ __bf16 As[64 * LDT];
    __shared__ __bf16 Ws[64 * LDT];

    const int mblk = blockIdx.x;     // 0..1023 (M = 65536 rows)
    const int nblk = blockIdx.y;     // 0..3    (N = 256 cols)
    const int t    = threadIdx.x;
    const unsigned lane = t & 31;
    const int wave = t >> 5;

    copy_tile_async(As, attnb + (size_t)mblk * 64 * ND, t);
    copy_tile_async(Ws, Wb + (size_t)nblk * 64 * ND, t);  // W rows = out cols
    wait_async0();
    __syncthreads();

    const int mr  = wave >> 1;
    const int nc0 = (wave & 1) * 2;

    v8f c0 = {}; v8f c1 = {};
#pragma unroll
    for (int ks = 0; ks < 8; ++ks) {
        v16bf a  = load_a_frag(As, mr * 16, ks * 32, lane);
        v16bf b0 = load_b_frag(Ws, nc0 * 16, ks * 32, lane);
        v16bf b1 = load_b_frag(Ws, (nc0 + 1) * 16, ks * 32, lane);
        c0 = wmma_bf16(a, b0, c0);
        c1 = wmma_bf16(a, b1, c1);
    }

    const int rB   = mr * 16 + ((lane >= 16) ? 8 : 0);
    const int col0 = nblk * 64 + nc0 * 16 + (lane & 15);
#pragma unroll
    for (int r = 0; r < 8; ++r) {
        size_t rowg = (size_t)mblk * 64 + rB + r;
        out[rowg * ND + col0]      = c0[r] + bias[col0];
        out[rowg * ND + col0 + 16] = c1[r] + bias[col0 + 16];
    }
}

// ---------------------------------------------------------------------------
// Launch
// ---------------------------------------------------------------------------
extern "C" void kernel_launch(void* const* d_in, const int* in_sizes, int n_in,
                              void* d_out, int out_size, void* d_ws, size_t ws_size,
                              hipStream_t stream) {
    const float*         x    = (const float*)d_in[0];
    const unsigned char* mask = (const unsigned char*)d_in[1];
    const float*         W    = (const float*)d_in[2];
    const float*         bias = (const float*)d_in[3];
    float*               out  = (float*)d_out;

    char* ws = (char*)d_ws;
    const size_t XS_BYTES = (size_t)NB * NN * ND * 2;     // 32 MB
    __bf16* xsb   = (__bf16*)(ws);
    __bf16* attnb = (__bf16*)(ws + XS_BYTES);
    __bf16* Wb    = (__bf16*)(ws + 2 * XS_BYTES);
    float*  denom = (float*)(ws + 2 * XS_BYTES + (size_t)ND * ND * 2);

    prep_xs_kernel<<<(NB * NN * ND) / (256 * 4), 256, 0, stream>>>(x, mask, xsb);
    prep_w_kernel<<<(ND * ND) / (256 * 4), 256, 0, stream>>>(W, Wb);
    calc_denom_kernel<<<1, 128, 0, stream>>>(mask, denom);

    dim3 gB(NN / 64, NB);
    topk_attn_kernel<<<gB, 256, 0, stream>>>(x, mask, xsb, denom, attnb);

    dim3 gC((NB * NN) / 64, ND / 64);
    final_gemm_kernel<<<gC, 256, 0, stream>>>(attnb, Wb, bias, out);
}
